// NTXentLoss_69449621176810
// MI455X (gfx1250) — matrix operand order repaired and checked by reading, not compiled, not run on hardware
//
#include <hip/hip_runtime.h>
#include <hip/hip_bf16.h>

typedef __attribute__((ext_vector_type(16))) _Float16 v16h;
typedef __attribute__((ext_vector_type(8)))  _Float16 v8h;
typedef __attribute__((ext_vector_type(8)))  float    v8f;
typedef __attribute__((ext_vector_type(4)))  int      v4i;

#define DDIM 256
#define TILE_ROWS 128   // rows per workgroup (8 waves x 16)
#define TILE_COLS 128   // columns staged in LDS per j-iteration
#define JSPLIT 4        // split of the column dimension across blocks

#if __has_builtin(__builtin_amdgcn_global_load_async_to_lds_b128)
#define HAVE_ASYNC_LDS 1
#else
#define HAVE_ASYNC_LDS 0
#endif

// Builtin signature (from hipcc diagnostic): (global v4i*, local v4i*, imm, imm)
typedef __attribute__((address_space(1))) v4i* gv4_ptr;
typedef __attribute__((address_space(3))) v4i* lv4_ptr;

__device__ __forceinline__ void wait_asynccnt0() {
#if __has_builtin(__builtin_amdgcn_s_wait_asynccnt)
  __builtin_amdgcn_s_wait_asynccnt(0);
#else
  asm volatile("s_wait_asynccnt 0x0" ::: "memory");
#endif
}

__device__ __forceinline__ const float* raw_row(const float* z1, const float* z2,
                                                const float* z3, const float* z4,
                                                int r, int B) {
  int v = r / B;
  const float* p = (v == 0) ? z1 : (v == 1) ? z2 : (v == 2) ? z3 : z4;
  return p + (size_t)(r - v * B) * DDIM;
}

__device__ __forceinline__ v16h cat16(v8h lo, v8h hi) {
  return __builtin_shufflevector(lo, hi, 0, 1, 2, 3, 4, 5, 6, 7,
                                         8, 9, 10, 11, 12, 13, 14, 15);
}

// ---------------------------------------------------------------------------
// Kernel 1: L2-normalize rows; emit fp16 Z (row-major [N,256]) + f32 inv-norms
// ---------------------------------------------------------------------------
__global__ void normalize_kernel(const float* __restrict__ z1, const float* __restrict__ z2,
                                 const float* __restrict__ z3, const float* __restrict__ z4,
                                 _Float16* __restrict__ Zh, float* __restrict__ invn, int B) {
  int lane = threadIdx.x & 31;
  int wave = threadIdx.x >> 5;
  int r = blockIdx.x * 8 + wave;
  const float* src = raw_row(z1, z2, z3, z4, r, B);
  float4 a = ((const float4*)src)[lane * 2 + 0];
  float4 b = ((const float4*)src)[lane * 2 + 1];
  float ss = a.x * a.x + a.y * a.y + a.z * a.z + a.w * a.w +
             b.x * b.x + b.y * b.y + b.z * b.z + b.w * b.w;
#pragma unroll
  for (int m = 1; m < 32; m <<= 1) ss += __shfl_xor(ss, m, 32);
  float inv = 1.0f / fmaxf(sqrtf(ss), 1e-12f);
  v8h h;
  h[0] = (_Float16)(a.x * inv); h[1] = (_Float16)(a.y * inv);
  h[2] = (_Float16)(a.z * inv); h[3] = (_Float16)(a.w * inv);
  h[4] = (_Float16)(b.x * inv); h[5] = (_Float16)(b.y * inv);
  h[6] = (_Float16)(b.z * inv); h[7] = (_Float16)(b.w * inv);
  *(v8h*)(Zh + (size_t)r * DDIM + lane * 8) = h;
  if (lane == 0) invn[r] = inv;
}

// ---------------------------------------------------------------------------
// Kernel 2: fused Gram + exp + row-sum over a j-range.  grid = (N/128, JSPLIT).
// Each wave owns a 16-row strip (A fragments in VGPRs, K=256); the block
// stages 128 columns of fp16 Z into LDS per iteration — via async global->LDS
// copies (ASYNCcnt) when the builtin exists, else a uint4 register copy.
// rowsum_part[jsplit][r] = sum_{j in range, j != r} exp(2 * z_r . z_j)
// ---------------------------------------------------------------------------
__global__ void gram_rowsum_kernel(const _Float16* __restrict__ Zh,
                                   float* __restrict__ rowsum_part, int N) {
  __shared__ __align__(16) _Float16 lds[TILE_COLS * DDIM];
  int lane = threadIdx.x & 31;
  int wave = threadIdx.x >> 5;
  int lmod = lane & 15;
  int halfsel = (lane < 16) ? 0 : 1;
  int rowbase = blockIdx.x * TILE_ROWS + wave * 16;

  // A fragments (ISA 16-bit A layout): lane<16 holds K {q..q+7, q+16..q+23}
  v16h afrag[8];
  {
    const _Float16* arow = Zh + (size_t)(rowbase + lmod) * DDIM;
#pragma unroll
    for (int k = 0; k < 8; ++k) {
      int q = k * 32 + halfsel * 8;
      v8h lo = *(const v8h*)(arow + q);
      v8h hi = *(const v8h*)(arow + q + 16);
      afrag[k] = cat16(lo, hi);
    }
  }

  float acc[8];
#pragma unroll
  for (int v = 0; v < 8; ++v) acc[v] = 0.0f;

  const int chunksPerSplit = N / (JSPLIT * TILE_COLS);
  const int jsplitBase = blockIdx.y * chunksPerSplit * TILE_COLS;

  for (int jc = 0; jc < chunksPerSplit; ++jc) {
    int jbase = jsplitBase + jc * TILE_COLS;
    __syncthreads();  // previous buffer fully consumed by all waves
    {
      const char* gsrc = (const char*)(Zh + (size_t)jbase * DDIM);
      char* lbase = (char*)lds;
      const int total = TILE_COLS * DDIM * 2 / 16;  // 4096 x 16B
      for (int c = threadIdx.x; c < total; c += blockDim.x) {
#if HAVE_ASYNC_LDS
        __builtin_amdgcn_global_load_async_to_lds_b128(
            (gv4_ptr)(gsrc + (size_t)c * 16), (lv4_ptr)(lbase + (size_t)c * 16),
            0, 0);
#else
        ((uint4*)lbase)[c] = ((const uint4*)gsrc)[c];
#endif
      }
#if HAVE_ASYNC_LDS
      wait_asynccnt0();
#endif
    }
    __syncthreads();

    for (int t = 0; t < TILE_COLS / 16; ++t) {
      // B fragments (match A layout): lane<16 holds K {q..q+7, q+16..q+23}
      const _Float16* brow = lds + (t * 16 + lmod) * DDIM;
      v8f c = {};
#pragma unroll
      for (int k = 0; k < 8; ++k) {
        int q = k * 32 + halfsel * 8;
        v8h lo = *(const v8h*)(brow + q);
        v8h hi = *(const v8h*)(brow + q + 16);
        v16h b = cat16(lo, hi);
        c = __builtin_amdgcn_wmma_f32_16x16x32_f16(
            /*neg_a=*/false, afrag[k], /*neg_b=*/false, b,
            /*c_mod=*/(short)0, c, /*reuse_a=*/false, /*reuse_b=*/false);
      }
      int col = jbase + t * 16 + lmod;
#pragma unroll
      for (int v = 0; v < 8; ++v) {
        int row = rowbase + v + halfsel * 8;
        acc[v] += (row == col) ? 0.0f : __expf(2.0f * c[v]);
      }
    }
  }

  // Sum over the j-range within each 16-lane half-group; store partial
#pragma unroll
  for (int v = 0; v < 8; ++v) {
    float s = acc[v];
    s += __shfl_xor(s, 1, 32);
    s += __shfl_xor(s, 2, 32);
    s += __shfl_xor(s, 4, 32);
    s += __shfl_xor(s, 8, 32);
    if (lmod == 0)
      rowsum_part[(size_t)blockIdx.y * N + rowbase + v + halfsel * 8] = s;
  }
}

// ---------------------------------------------------------------------------
// Kernel 3: per-row loss = -pos(r) + log(denom + eps); positive partner of row
// r is (r + B) mod N, computed in f32 from the raw inputs + inverse norms.
// ---------------------------------------------------------------------------
__global__ void finalize_kernel(const float* __restrict__ z1, const float* __restrict__ z2,
                                const float* __restrict__ z3, const float* __restrict__ z4,
                                const float* __restrict__ invn,
                                const float* __restrict__ rowsum_part,
                                float* __restrict__ rowloss, int B, int N) {
  int lane = threadIdx.x & 31;
  int wave = threadIdx.x >> 5;
  int r = blockIdx.x * 8 + wave;
  int p = r + B;
  if (p >= N) p -= N;
  const float* ra = raw_row(z1, z2, z3, z4, r, B);
  const float* rb = raw_row(z1, z2, z3, z4, p, B);
  float4 a0 = ((const float4*)ra)[lane * 2 + 0];
  float4 a1 = ((const float4*)ra)[lane * 2 + 1];
  float4 b0 = ((const float4*)rb)[lane * 2 + 0];
  float4 b1 = ((const float4*)rb)[lane * 2 + 1];
  float dot = a0.x * b0.x + a0.y * b0.y + a0.z * b0.z + a0.w * b0.w +
              a1.x * b1.x + a1.y * b1.y + a1.z * b1.z + a1.w * b1.w;
#pragma unroll
  for (int m = 1; m < 32; m <<= 1) dot += __shfl_xor(dot, m, 32);
  if (lane == 0) {
    float denom = 0.0f;
#pragma unroll
    for (int s = 0; s < JSPLIT; ++s) denom += rowsum_part[(size_t)s * N + r];
    float pos = 2.0f * dot * invn[r] * invn[p];  // sim / T, T = 0.5
    rowloss[r] = -pos + logf(denom + 1e-12f);
  }
}

// ---------------------------------------------------------------------------
// Kernel 4: mean over N row losses -> scalar
// ---------------------------------------------------------------------------
__global__ void mean_kernel(const float* __restrict__ rowloss, float* __restrict__ out, int N) {
  __shared__ float s[256];
  float acc = 0.0f;
  for (int i = threadIdx.x; i < N; i += 256) acc += rowloss[i];
  s[threadIdx.x] = acc;
  __syncthreads();
  for (int k = 128; k > 0; k >>= 1) {
    if ((int)threadIdx.x < k) s[threadIdx.x] += s[threadIdx.x + k];
    __syncthreads();
  }
  if (threadIdx.x == 0) out[0] = s[0] / (float)N;
}

extern "C" void kernel_launch(void* const* d_in, const int* in_sizes, int n_in,
                              void* d_out, int out_size, void* d_ws, size_t ws_size,
                              hipStream_t stream) {
  const float* z1 = (const float*)d_in[0];
  const float* z2 = (const float*)d_in[1];
  const float* z3 = (const float*)d_in[2];
  const float* z4 = (const float*)d_in[3];
  const int B = in_sizes[0] / DDIM;  // 4096
  const int N = 4 * B;               // 16384

  char* ws = (char*)d_ws;
  _Float16* Zh       = (_Float16*)ws;  // N*D*2 bytes = 8 MB
  size_t off         = (size_t)N * DDIM * sizeof(_Float16);
  float* invn        = (float*)(ws + off);  off += (size_t)N * sizeof(float);
  float* rowsum_part = (float*)(ws + off);  off += (size_t)N * JSPLIT * sizeof(float);
  float* rowloss     = (float*)(ws + off);

  normalize_kernel<<<N / 8, 256, 0, stream>>>(z1, z2, z3, z4, Zh, invn, B);
  dim3 grid(N / TILE_ROWS, JSPLIT);
  gram_rowsum_kernel<<<grid, 256, 0, stream>>>(Zh, rowsum_part, N);
  finalize_kernel<<<N / 8, 256, 0, stream>>>(z1, z2, z3, z4, invn, rowsum_part, rowloss, B, N);
  mean_kernel<<<1, 256, 0, stream>>>(rowloss, (float*)d_out, N);
}
